// AttentionBlock_70574902607913
// MI455X (gfx1250) — matrix operand order repaired and checked by reading, not compiled, not run on hardware
//
#include <hip/hip_runtime.h>

typedef __attribute__((ext_vector_type(16))) _Float16 v16h;
typedef __attribute__((ext_vector_type(8)))  _Float16 v8h;
typedef __attribute__((ext_vector_type(8)))  float    v8f;

static constexpr int Bn  = 8;
static constexpr int Cn  = 256;
static constexpr int Nn  = 4096;      // H*W
static constexpr int Gn  = 32;
static constexpr int CPG = Cn / Gn;   // 8 channels per group

__device__ __forceinline__ v8f wmma16(v16h a, v16h b, v8f c) {
  return __builtin_amdgcn_wmma_f32_16x16x32_f16(false, a, false, b, (short)0, c,
                                                false, false);
}

// Intra-wave LDS ordering point (DS ops from one wave complete in order).
__device__ __forceinline__ void lds_fence_wave() {
  asm volatile("s_wait_dscnt 0" ::: "memory");
}

// CDNA5 async global->LDS copy, 16B per lane, tracked by ASYNCcnt.
__device__ __forceinline__ void async_load_b128(uint32_t lds_off,
                                                const void* gaddr) {
  asm volatile("global_load_async_to_lds_b128 %0, %1, off"
               :: "v"(lds_off), "v"(gaddr) : "memory");
}
__device__ __forceinline__ void wait_async() {
  asm volatile("s_wait_asynccnt 0" ::: "memory");
}

// A fragment (16 rows x 32 K, f16) from a row-major [16 x >=32] tile.
// lane l: row = l%16, half = l/16; element e -> col 16*half + e
__device__ __forceinline__ v16h load_frag_a(const _Float16* __restrict__ base,
                                            int row_stride, int lane) {
  int r = lane & 15, hf = lane >> 4;
  const _Float16* p = base + (size_t)r * row_stride + hf * 16;
  v8h lo = *(const v8h*)(p);
  v8h hi = *(const v8h*)(p + 8);
  v16h out;
#pragma unroll
  for (int e = 0; e < 8; ++e) { out[e] = lo[e]; out[e + 8] = hi[e]; }
  return out;
}

// B fragment (32 K x 16 cols, f16) loaded from B^T stored row-major [16 x >=32].
// lane l: col = l%16; element e -> K = 16*(l/16) + e  (16 contiguous halves)
__device__ __forceinline__ v16h load_frag_b(const _Float16* __restrict__ baseT,
                                            int row_stride, int lane) {
  int cl = lane & 15, hf = lane >> 4;
  return *(const v16h*)(baseT + (size_t)cl * row_stride + hf * 16);
}

// ---------------- K0: fp32 -> f16 weight conversion ----------------
__global__ __launch_bounds__(256) void cvt_w_kernel(
    const float* __restrict__ wq, const float* __restrict__ wp,
    _Float16* __restrict__ wq16, _Float16* __restrict__ wp16) {
  int i = blockIdx.x * 256 + threadIdx.x;
  if (i < 3 * Cn * Cn) wq16[i] = (_Float16)wq[i];
  if (i < Cn * Cn)     wp16[i] = (_Float16)wp[i];
}

// ---------------- K1: GroupNorm + transpose to (B, N, C) f16 ----------------
__global__ __launch_bounds__(256) void gn_kernel(
    const float* __restrict__ x, const float* __restrict__ gamma,
    const float* __restrict__ beta, _Float16* __restrict__ h) {
  int b = blockIdx.x / Gn, g = blockIdx.x % Gn;
  int c0 = g * CPG;
  const float* xb = x + ((size_t)(b * Cn + c0)) * Nn;
  float s = 0.f, s2 = 0.f;
  for (int idx = threadIdx.x; idx < CPG * Nn; idx += 256) {
    float v = xb[idx];
    s += v; s2 += v * v;
  }
  __shared__ float red0[256], red1[256];
  red0[threadIdx.x] = s; red1[threadIdx.x] = s2;
  __syncthreads();
  for (int off = 128; off > 0; off >>= 1) {
    if ((int)threadIdx.x < off) {
      red0[threadIdx.x] += red0[threadIdx.x + off];
      red1[threadIdx.x] += red1[threadIdx.x + off];
    }
    __syncthreads();
  }
  const float inv_n = 1.f / (float)(CPG * Nn);
  float mean = red0[0] * inv_n;
  float var  = red1[0] * inv_n - mean * mean;
  float rstd = rsqrtf(var + 1e-5f);
  float gm[CPG], bt[CPG];
#pragma unroll
  for (int ch = 0; ch < CPG; ++ch) {
    gm[ch] = gamma[c0 + ch] * rstd;
    bt[ch] = beta[c0 + ch] - mean * gm[ch];
  }
  for (int n = threadIdx.x; n < Nn; n += 256) {
    v8h o;
#pragma unroll
    for (int ch = 0; ch < CPG; ++ch)
      o[ch] = (_Float16)(xb[(size_t)ch * Nn + n] * gm[ch] + bt[ch]);
    *(v8h*)(h + ((size_t)(b * Nn + n)) * Cn + c0) = o;  // 16B aligned
  }
}

// ---------------- K2: QKV GEMM ----------------
// out[b,o,n] = sum_c h[b,n,c] * w[o,c] + bias[o]
// D tile: M = n (16), N = o (16); wave does 16n x 64o. Destination (q/k/vT)
// is wave-uniform since each 64-wide o-group stays inside one of the thirds.
__global__ __launch_bounds__(256) void qkv_kernel(
    const _Float16* __restrict__ h, const _Float16* __restrict__ w16,
    const float* __restrict__ bias, _Float16* __restrict__ q,
    _Float16* __restrict__ k, _Float16* __restrict__ vT) {
  int lane = threadIdx.x & 31;
  int wid = (blockIdx.x * 256 + threadIdx.x) >> 5;
  int og = wid % 12;  int t = wid / 12;
  int nt = t % (Nn / 16);  int b = t / (Nn / 16);
  int n0 = nt * 16;
  const _Float16* hbase = h + ((size_t)b * Nn + n0) * Cn;
  v8f acc[4] = {};
#pragma unroll
  for (int k0 = 0; k0 < Cn; k0 += 32) {
    v16h a = load_frag_a(hbase + k0, Cn, lane);
    v16h b0 = load_frag_b(w16 + (size_t)(og * 64 +  0) * Cn + k0, Cn, lane);
    v16h b1 = load_frag_b(w16 + (size_t)(og * 64 + 16) * Cn + k0, Cn, lane);
    v16h b2 = load_frag_b(w16 + (size_t)(og * 64 + 32) * Cn + k0, Cn, lane);
    v16h b3 = load_frag_b(w16 + (size_t)(og * 64 + 48) * Cn + k0, Cn, lane);
    acc[0] = wmma16(a, b0, acc[0]);
    acc[1] = wmma16(a, b1, acc[1]);
    acc[2] = wmma16(a, b2, acc[2]);
    acc[3] = wmma16(a, b3, acc[3]);
  }
  int colh = lane >> 4, coln = lane & 15;
  if (og < 8) {                      // Q or K: (B, N, C) f16, row-major
    _Float16* dst = (og < 4) ? q : k;
    int obase = (og & 3) * 64;
#pragma unroll
    for (int j = 0; j < 4; ++j) {
      int o = obase + j * 16 + coln;
      float bo = bias[(og < 4 ? 0 : Cn) + o];
#pragma unroll
      for (int i = 0; i < 8; ++i) {
        int n = n0 + colh * 8 + i;
        dst[((size_t)b * Nn + n) * Cn + o] = (_Float16)(acc[j][i] + bo);
      }
    }
  } else {                           // V: transposed to (B, C, N) f16
    int obase = (og - 8) * 64;
#pragma unroll
    for (int j = 0; j < 4; ++j) {
      int o = obase + j * 16 + coln;
      float bo = bias[2 * Cn + o];
#pragma unroll
      for (int i = 0; i < 8; ++i) {
        int n = n0 + colh * 8 + i;
        vT[((size_t)b * Cn + o) * Nn + n] = (_Float16)(acc[j][i] + bo);
      }
    }
  }
}

// ---------------- K3: flash attention ----------------
// Block = 4 waves (same batch, consecutive query tiles). K/V chunks are
// staged once per block in LDS via async global->LDS copies, double-buffered;
// all 4 waves consume the shared tiles. One wave owns 16 query rows.
__global__ __launch_bounds__(128) void attn_kernel(
    const _Float16* __restrict__ q, const _Float16* __restrict__ k,
    const _Float16* __restrict__ vT, _Float16* __restrict__ ao) {
  __shared__ __align__(16) _Float16 Kbuf[2][32][264];   // 32 keys x 256c (+pad)
  __shared__ __align__(16) _Float16 Vbuf[2][256][40];   // 256c x 32 keys (+pad)
  __shared__ __align__(16) _Float16 Pbuf[4][16][40];    // per-wave P tiles

  int lane = threadIdx.x & 31;
  int w = threadIdx.x >> 5;
  int wid = blockIdx.x * 4 + w;
  int qt = wid % (Nn / 16);  int b = wid / (Nn / 16);
  int n0 = qt * 16;
  const _Float16* qbase = q + ((size_t)b * Nn + n0) * Cn;
  const _Float16* kbase = k + (size_t)b * Nn * Cn;
  const _Float16* vbase = vT + (size_t)b * Cn * Nn;

  v16h qf[8];                       // resident Q fragments (K = 256)
#pragma unroll
  for (int kk = 0; kk < 8; ++kk) qf[kk] = load_frag_a(qbase + kk * 32, Cn, lane);

  v8f acc[16] = {};                 // O accumulators, 16 c-tiles
  float rmax[8], rsum[8];
#pragma unroll
  for (int j = 0; j < 8; ++j) { rmax[j] = -3.0e38f; rsum[j] = 0.f; }
  int colh = lane >> 4, coln = lane & 15;
  const float SCL = 0.0625f * 1.4426950408889634f;  // 1/sqrt(C) * log2(e)

  // async-stage one 32-key chunk (K tile + V tile) into buffer nb
  auto issue_chunk = [&](int m0, int nb) {
    uint32_t kloff = (uint32_t)(size_t)&Kbuf[nb][0][0];
    uint32_t vloff = (uint32_t)(size_t)&Vbuf[nb][0][0];
#pragma unroll
    for (int i = 0; i < 8; ++i) {                 // K: 8 of 32 rows per wave
      int row = w * 8 + i;
      const _Float16* g = kbase + (size_t)(m0 + row) * Cn + lane * 8;
      async_load_b128(kloff + (uint32_t)(row * 264 + lane * 8) * 2, g);
    }
#pragma unroll
    for (int i = 0; i < 8; ++i) {                 // V: 64 of 256 rows per wave
      int row = w * 64 + i * 8 + (lane >> 2);
      const _Float16* g = vbase + (size_t)row * Nn + m0 + (lane & 3) * 8;
      async_load_b128(vloff + (uint32_t)(row * 40 + (lane & 3) * 8) * 2, g);
    }
  };

  issue_chunk(0, 0);
  wait_async();
  __syncthreads();

#pragma unroll 1
  for (int ch = 0; ch < Nn / 32; ++ch) {
    int cur = ch & 1;
    if (ch + 1 < Nn / 32) issue_chunk((ch + 1) * 32, cur ^ 1);

    // S = Q x K^T for 2 key tiles of this chunk, from LDS
    v8f s0 = {}, s1 = {};
#pragma unroll
    for (int kk = 0; kk < 8; ++kk) {
      v16h b0 = *(const v16h*)&Kbuf[cur][coln][kk * 32 + colh * 16];
      v16h b1 = *(const v16h*)&Kbuf[cur][16 + coln][kk * 32 + colh * 16];
      s0 = wmma16(qf[kk], b0, s0);
      s1 = wmma16(qf[kk], b1, s1);
    }
    // online softmax in exp2 domain; row M = 8*(lane/16)+j across 16-lane half
#pragma unroll
    for (int j = 0; j < 8; ++j) {
      float v0 = s0[j] * SCL;
      float v1 = s1[j] * SCL;
      float cm = fmaxf(v0, v1);
#pragma unroll
      for (int mm = 1; mm < 16; mm <<= 1) cm = fmaxf(cm, __shfl_xor(cm, mm, 32));
      float nm = fmaxf(rmax[j], cm);
      float alpha = exp2f(rmax[j] - nm);
      rmax[j] = nm;
      float p0 = exp2f(v0 - nm), p1 = exp2f(v1 - nm);
      float ps = p0 + p1;
#pragma unroll
      for (int mm = 1; mm < 16; mm <<= 1) ps += __shfl_xor(ps, mm, 32);
      rsum[j] = rsum[j] * alpha + ps;
#pragma unroll
      for (int f = 0; f < 16; ++f) acc[f][j] *= alpha;
      int row = colh * 8 + j;
      Pbuf[w][row][coln]      = (_Float16)p0;
      Pbuf[w][row][16 + coln] = (_Float16)p1;
    }
    lds_fence_wave();               // wave-local P store->load ordering
    v16h pf;                        // P tile: C/D layout -> A layout via LDS
    {
      int r = lane & 15, hf = lane >> 4;
      v8h lo = *(const v8h*)&Pbuf[w][r][16 * hf];
      v8h hi = *(const v8h*)&Pbuf[w][r][16 * hf + 8];
#pragma unroll
      for (int e = 0; e < 8; ++e) { pf[e] = lo[e]; pf[e + 8] = hi[e]; }
    }
    // O += P x V, V fragments from LDS
#pragma unroll
    for (int f = 0; f < 16; ++f) {
      v16h vf = *(const v16h*)&Vbuf[cur][f * 16 + coln][colh * 16];
      acc[f] = wmma16(pf, vf, acc[f]);
    }

    wait_async();                   // next chunk landed (this wave's slice)
    __syncthreads();                // all waves done reading cur / writing next
  }
  // normalize and store attention output as (B, N, C) f16
#pragma unroll
  for (int j = 0; j < 8; ++j) {
    float inv = 1.f / rsum[j];
    int n = n0 + colh * 8 + j;
#pragma unroll
    for (int f = 0; f < 16; ++f)
      ao[((size_t)b * Nn + n) * Cn + f * 16 + coln] = (_Float16)(acc[f][j] * inv);
  }
}

// ---------------- K4: proj GEMM + bias + residual ----------------
// D tile: M = o (16), N = n (16); wave does 16o x 64n, w_proj A-frags resident.
__global__ __launch_bounds__(256) void proj_kernel(
    const _Float16* __restrict__ ao, const _Float16* __restrict__ wp16,
    const float* __restrict__ bias, const float* __restrict__ x,
    float* __restrict__ out) {
  int lane = threadIdx.x & 31;
  int wid = (blockIdx.x * 256 + threadIdx.x) >> 5;
  int ng = wid % 64;  int t = wid / 64;
  int ot = t % (Cn / 16);  int b = t / (Cn / 16);
  int o0 = ot * 16;
  v16h af[8];
#pragma unroll
  for (int kk = 0; kk < 8; ++kk)
    af[kk] = load_frag_a(wp16 + (size_t)o0 * Cn + kk * 32, Cn, lane);
  v8f acc[4] = {};
#pragma unroll
  for (int kk = 0; kk < 8; ++kk) {
    v16h bf0 = load_frag_b(ao + ((size_t)b * Nn + ng * 64 +  0) * Cn + kk * 32, Cn, lane);
    v16h bf1 = load_frag_b(ao + ((size_t)b * Nn + ng * 64 + 16) * Cn + kk * 32, Cn, lane);
    v16h bf2 = load_frag_b(ao + ((size_t)b * Nn + ng * 64 + 32) * Cn + kk * 32, Cn, lane);
    v16h bf3 = load_frag_b(ao + ((size_t)b * Nn + ng * 64 + 48) * Cn + kk * 32, Cn, lane);
    acc[0] = wmma16(af[kk], bf0, acc[0]);
    acc[1] = wmma16(af[kk], bf1, acc[1]);
    acc[2] = wmma16(af[kk], bf2, acc[2]);
    acc[3] = wmma16(af[kk], bf3, acc[3]);
  }
  int colh = lane >> 4, coln = lane & 15;
#pragma unroll
  for (int j = 0; j < 4; ++j) {
    int n = ng * 64 + j * 16 + coln;
#pragma unroll
    for (int i = 0; i < 8; ++i) {
      int o = o0 + colh * 8 + i;
      size_t idx = ((size_t)b * Cn + o) * Nn + n;
      out[idx] = x[idx] + bias[o] + acc[j][i];
    }
  }
}

extern "C" void kernel_launch(void* const* d_in, const int* in_sizes, int n_in,
                              void* d_out, int out_size, void* d_ws, size_t ws_size,
                              hipStream_t stream) {
  const float* x     = (const float*)d_in[0];
  const float* gamma = (const float*)d_in[1];
  const float* beta  = (const float*)d_in[2];
  const float* w_qkv = (const float*)d_in[3];
  const float* b_qkv = (const float*)d_in[4];
  const float* w_prj = (const float*)d_in[5];
  const float* b_prj = (const float*)d_in[6];
  float* out = (float*)d_out;

  // workspace layout (f16 tensors), all offsets 256B aligned
  char* ws = (char*)d_ws;
  const size_t SZ = (size_t)Bn * Nn * Cn * sizeof(_Float16);  // 16 MB each
  _Float16* h16  = (_Float16*)(ws);
  _Float16* q16  = (_Float16*)(ws + SZ);
  _Float16* k16  = (_Float16*)(ws + 2 * SZ);
  _Float16* vT16 = (_Float16*)(ws + 3 * SZ);
  _Float16* ao16 = (_Float16*)(ws + 4 * SZ);
  _Float16* wq16 = (_Float16*)(ws + 5 * SZ);
  _Float16* wp16 = (_Float16*)(ws + 5 * SZ + (size_t)3 * Cn * Cn * sizeof(_Float16));

  cvt_w_kernel<<<(3 * Cn * Cn + 255) / 256, 256, 0, stream>>>(w_qkv, w_prj, wq16, wp16);
  gn_kernel<<<Bn * Gn, 256, 0, stream>>>(x, gamma, beta, h16);
  // waves = B * (N/16) * 12 = 24576 -> 3072 blocks of 8 waves
  qkv_kernel<<<(Bn * (Nn / 16) * 12) / 8, 256, 0, stream>>>(h16, wq16, b_qkv,
                                                            q16, k16, vT16);
  // waves = B * (N/16) = 2048 -> 512 blocks of 4 waves
  attn_kernel<<<(Bn * (Nn / 16)) / 4, 128, 0, stream>>>(q16, k16, vT16, ao16);
  // waves = B * (C/16) * (N/64) = 8192 -> 1024 blocks of 8 waves
  proj_kernel<<<(Bn * (Cn / 16) * 64) / 8, 256, 0, stream>>>(ao16, wp16, b_prj,
                                                             x, out);
}
